// SHA_24326694764740
// MI455X (gfx1250) — matrix operand order repaired and checked by reading, not compiled, not run on hardware
//
#include <hip/hip_runtime.h>
#include <hip/hip_bf16.h>

typedef _Float16 h16 __attribute__((ext_vector_type(16)));
typedef _Float16 h8  __attribute__((ext_vector_type(8)));
typedef float    f8  __attribute__((ext_vector_type(8)));
typedef unsigned int u32x4 __attribute__((ext_vector_type(4)));
typedef int          i32x8 __attribute__((ext_vector_type(8)));
typedef int          i32x4 __attribute__((ext_vector_type(4)));

#define DMODEL 1024
#define DK     64
#define BATCH  4
#define SEQ    4096
#define ROWS   (BATCH * SEQ)   // 16384

// ---------------------------------------------------------------------------
// WMMA helpers (gfx1250: V_WMMA_F32_16X16X32_F16, wave32)
// A 16x32 f16 layout: lane<16 -> row=lane,   K chunks {0..7, 16..23}
//                     lane>=16 -> row=lane-16, K chunks {8..15, 24..31}
// B 32x16 f16 layout: lane<16 -> col=lane,   K {0..7, 16..23}; lane>=16 -> K {8..15, 24..31}
// C/D 16x16 f32:      lane -> N=lane%16, vgpr v -> M = v + (lane>=16 ? 8 : 0)
// ---------------------------------------------------------------------------

static __device__ __forceinline__ f8 wmma32(h16 a, h16 b, f8 c) {
  return __builtin_amdgcn_wmma_f32_16x16x32_f16(false, a, false, b,
                                                (short)0, c, false, false);
}

static __device__ __forceinline__ h16 load_a_f16(const _Float16* src, int ld, int lane) {
  int m  = lane & 15;
  int ko = (lane >= 16) ? 8 : 0;
  const _Float16* p = src + (size_t)m * ld + ko;
  h8 c0 = *(const h8*)(p);
  h8 c1 = *(const h8*)(p + 16);
  h16 a;
#pragma unroll
  for (int i = 0; i < 8; i++) { a[i] = c0[i]; a[8 + i] = c1[i]; }
  return a;
}

static __device__ __forceinline__ h16 load_a_f32(const float* src, int ld, int lane) {
  int m  = lane & 15;
  int ko = (lane >= 16) ? 8 : 0;
  const float* p = src + (size_t)m * ld + ko;
  f8 c0 = *(const f8*)(p);
  f8 c1 = *(const f8*)(p + 16);
  h16 a;
#pragma unroll
  for (int i = 0; i < 8; i++) { a[i] = (_Float16)c0[i]; a[8 + i] = (_Float16)c1[i]; }
  return a;
}

static __device__ __forceinline__ h16 load_b_f16(const _Float16* src, int ld, int lane) {
  int n  = lane & 15;
  int ko = (lane >= 16) ? 8 : 0;
  const _Float16* p = src + (size_t)n * ld + ko;
  h8 c0 = *(const h8*)(p);
  h8 c1 = *(const h8*)(p + 16);
  h16 b;
#pragma unroll
  for (int i = 0; i < 8; i++) { b[i] = c0[i]; b[8 + i] = c1[i]; }
  return b;
}

// ---------------------------------------------------------------------------
// TDM: 2D tile load Global -> LDS (D# per CDNA5 ISA)
// data_size = 1 (2-byte elements). Issued once per wave (EXEC ignored).
// This toolchain's builtin takes 6 args: (g0, g1, g2, g3, g4, cpol).
// ---------------------------------------------------------------------------
static __device__ __forceinline__ void tdm_load_2d(
    unsigned lds_off, unsigned long long gaddr,
    unsigned tensor_d0, unsigned tensor_d1, unsigned long long stride0,
    unsigned tile_d0, unsigned tile_d1) {
  u32x4 g0;
  g0[0] = 1u;                                                   // count=1, user D#
  g0[1] = lds_off;                                              // lds_addr (bytes)
  g0[2] = (unsigned)gaddr;                                      // global_addr[31:0]
  g0[3] = ((unsigned)(gaddr >> 32) & 0x01FFFFFFu) | 0x80000000u; // addr[56:32]|type=2
  i32x8 g1;
  g1[0] = (int)(1u << 16);                                      // data_size=1 (2B)
  g1[1] = (int)((tensor_d0 & 0xFFFFu) << 16);                   // tensor_dim0 lo
  g1[2] = (int)((tensor_d0 >> 16) | ((tensor_d1 & 0xFFFFu) << 16));
  g1[3] = (int)((tensor_d1 >> 16) | (tile_d0 << 16));           // tile_dim0
  g1[4] = (int)tile_d1;                                         // tile_dim1 (dim2=0)
  g1[5] = (int)(unsigned)(stride0 & 0xFFFFFFFFu);               // dim0 stride lo
  g1[6] = (int)(unsigned)((stride0 >> 32) & 0xFFFFu);           // stride hi (dim1 str=0)
  g1[7] = 0;
  i32x4 g2 = {0, 0, 0, 0};
  i32x4 g3 = {0, 0, 0, 0};
  i32x8 g4 = {0, 0, 0, 0, 0, 0, 0, 0};
  __builtin_amdgcn_tensor_load_to_lds(g0, g1, g2, g3, g4, 0);
}

// ---------------------------------------------------------------------------
// Kernel 0: W (1024x64 f32) -> WT (64x1024 f16), three weights
// ---------------------------------------------------------------------------
__global__ void sha_wprep(const float* __restrict__ Wq,
                          const float* __restrict__ Wk,
                          const float* __restrict__ Wv,
                          _Float16* __restrict__ WT) {
  int w = blockIdx.y;
  const float* W = (w == 0) ? Wq : (w == 1) ? Wk : Wv;
  _Float16* dst = WT + (size_t)w * DK * DMODEL;
  int idx = blockIdx.x * blockDim.x + threadIdx.x;
  if (idx < DMODEL * DK) {
    int k = idx / DK, n = idx % DK;
    dst[(size_t)n * DMODEL + k] = (_Float16)W[idx];
  }
}

// ---------------------------------------------------------------------------
// Kernel 1: projection GEMM  [ROWS x 1024] x [1024 x 64] (+bias, *scale)
// ---------------------------------------------------------------------------
__global__ __launch_bounds__(128) void sha_proj(
    const float* __restrict__ X, const _Float16* __restrict__ WT,
    const float* __restrict__ bias,
    _Float16* __restrict__ out_qk, _Float16* __restrict__ out_vT,
    int transpose, float out_scale) {
  int lane = threadIdx.x & 31;
  int wave = threadIdx.x >> 5;
  int m0 = (blockIdx.x * 4 + wave) * 16;

  f8 acc[4] = {};
  for (int kc = 0; kc < DMODEL; kc += 32) {
    h16 a = load_a_f32(X + (size_t)m0 * DMODEL + kc, DMODEL, lane);
    h16 bt[4];
#pragma unroll
    for (int nt = 0; nt < 4; nt++)
      bt[nt] = load_b_f16(WT + (size_t)(nt * 16) * DMODEL + kc, DMODEL, lane);
#pragma unroll
    for (int nt = 0; nt < 4; nt++)
      acc[nt] = wmma32(a, bt[nt], acc[nt]);
  }

  int n = lane & 15;
  int rbase = (lane >= 16) ? 8 : 0;
  if (!transpose) {
#pragma unroll
    for (int nt = 0; nt < 4; nt++) {
      int col = nt * 16 + n;
      float bcol = bias[col];
#pragma unroll
      for (int v = 0; v < 8; v++) {
        int row = m0 + rbase + v;
        out_qk[(size_t)row * DK + col] = (_Float16)((acc[nt][v] + bcol) * out_scale);
      }
    }
  } else {
    int b  = m0 >> 12;
    int l0 = (m0 & (SEQ - 1)) + rbase;
#pragma unroll
    for (int nt = 0; nt < 4; nt++) {
      int col = nt * 16 + n;
      float bcol = bias[col];
      h8 pack;
#pragma unroll
      for (int v = 0; v < 8; v++)
        pack[v] = (_Float16)((acc[nt][v] + bcol) * out_scale);
      *(h8*)(out_vT + (size_t)b * DK * SEQ + (size_t)col * SEQ + l0) = pack;
    }
  }
}

// ---------------------------------------------------------------------------
// Kernel 2: causal flash attention.
// 4 waves/block = 64 query rows. K/V tiles staged Global->LDS by the TDM
// (wave 0 issues, TENSORcnt-tracked, double buffered); all waves consume.
// All waves share the block-uniform key loop; early waves' out-of-range
// blocks are fully masked (exp -> 0), so they are mathematically inert.
// q16 already scaled by 1/sqrt(DK).
// ---------------------------------------------------------------------------
__global__ __launch_bounds__(128) void sha_attn(
    const _Float16* __restrict__ q16, const _Float16* __restrict__ k16,
    const _Float16* __restrict__ vT16, float* __restrict__ out) {
  __shared__ _Float16 klds[2][32][DK];   // [buf][key][dk]   4KB each
  __shared__ _Float16 vlds[2][DK][32];   // [buf][dk][key]   4KB each
  __shared__ _Float16 pbuf[4][16][48];   // per-wave P transpose tile

  int lane = threadIdx.x & 31;
  int wave = threadIdx.x >> 5;
  int rgblk = blockIdx.x * 64;           // first global row of this block
  int b     = rgblk >> 12;
  int q0blk = rgblk & (SEQ - 1);
  int rg0   = rgblk + wave * 16;         // this wave's global row base
  int q0    = q0blk + wave * 16;         // this wave's row base within batch

  const _Float16* kbase = k16  + (size_t)b * SEQ * DK;
  const _Float16* vbase = vT16 + (size_t)b * DK * SEQ;

  // Q A-operands (constant over the key loop)
  h16 aq0 = load_a_f16(q16 + (size_t)rg0 * DK + 0,  DK, lane);
  h16 aq1 = load_a_f16(q16 + (size_t)rg0 * DK + 32, DK, lane);

  f8 acc[4] = {};
  float mrow[8], lrow[8];
#pragma unroll
  for (int v = 0; v < 8; v++) { mrow[v] = -3.0e30f; lrow[v] = 0.0f; }

  int n = lane & 15;
  int rbase = (lane >= 16) ? 8 : 0;
  int kbEnd = q0blk + 64;                // uniform across the block

  unsigned koff[2] = { (unsigned)(unsigned long long)&klds[0][0][0],
                       (unsigned)(unsigned long long)&klds[1][0][0] };
  unsigned voff[2] = { (unsigned)(unsigned long long)&vlds[0][0][0],
                       (unsigned)(unsigned long long)&vlds[1][0][0] };

  if (wave == 0) {   // prologue DMA: key block 0 into buffer 0
    tdm_load_2d(koff[0], (unsigned long long)kbase, DK, SEQ, DK, DK, 32);
    tdm_load_2d(voff[0], (unsigned long long)vbase, SEQ, DK, SEQ, 32, DK);
  }

  int ibuf = 0;
  for (int kb = 0; kb < kbEnd; kb += 32, ibuf ^= 1) {
    if (wave == 0) {
      if (kb + 32 < kbEnd) {             // issue next block's DMA, then drain
        int nb = ibuf ^ 1;               // current buffer's two loads
        tdm_load_2d(koff[nb],
                    (unsigned long long)kbase + (size_t)(kb + 32) * DK * 2,
                    DK, SEQ, DK, DK, 32);
        tdm_load_2d(voff[nb],
                    (unsigned long long)vbase + (size_t)(kb + 32) * 2,
                    SEQ, DK, SEQ, 32, DK);
        __builtin_amdgcn_s_wait_tensorcnt(2);
      } else {
        __builtin_amdgcn_s_wait_tensorcnt(0);
      }
    }
    __syncthreads();                     // K/V tiles for kb are in LDS[ibuf]

    // S tiles from LDS: keys [kb,kb+16) and [kb+16,kb+32), K-dim = DK
    h16 bk0 = load_b_f16(&klds[ibuf][0][0],   DK, lane);
    h16 bk1 = load_b_f16(&klds[ibuf][0][32],  DK, lane);
    h16 bk2 = load_b_f16(&klds[ibuf][16][0],  DK, lane);
    h16 bk3 = load_b_f16(&klds[ibuf][16][32], DK, lane);
    f8 s0 = {}, s1 = {};
    s0 = wmma32(aq0, bk0, s0);
    s0 = wmma32(aq1, bk1, s0);
    s1 = wmma32(aq0, bk2, s1);
    s1 = wmma32(aq1, bk3, s1);

    if (kb + 31 > q0) {                  // diagonal / fully-masked blocks
#pragma unroll
      for (int v = 0; v < 8; v++) {
        int mg = q0 + rbase + v;
        if (kb + n      > mg) s0[v] = -3.0e30f;
        if (kb + 16 + n > mg) s1[v] = -3.0e30f;
      }
    }

    // online softmax (keys live on lanes: 16-lane butterflies)
    float scl[8];
    f8 p0, p1;
#pragma unroll
    for (int v = 0; v < 8; v++) {
      float x = fmaxf(s0[v], s1[v]);
      x = fmaxf(x, __shfl_xor(x, 1, 32));
      x = fmaxf(x, __shfl_xor(x, 2, 32));
      x = fmaxf(x, __shfl_xor(x, 4, 32));
      x = fmaxf(x, __shfl_xor(x, 8, 32));
      float mn = fmaxf(mrow[v], x);
      scl[v] = __expf(mrow[v] - mn);
      float e0 = __expf(s0[v] - mn);
      float e1 = __expf(s1[v] - mn);
      p0[v] = e0; p1[v] = e1;
      float s = e0 + e1;
      s += __shfl_xor(s, 1, 32);
      s += __shfl_xor(s, 2, 32);
      s += __shfl_xor(s, 4, 32);
      s += __shfl_xor(s, 8, 32);
      lrow[v] = lrow[v] * scl[v] + s;
      mrow[v] = mn;
    }
#pragma unroll
    for (int nt = 0; nt < 4; nt++)
#pragma unroll
      for (int v = 0; v < 8; v++) acc[nt][v] *= scl[v];

    // C-layout P -> LDS (row,key) -> A-layout (per-wave tile; same-wave LDS
    // ops are in-order, no barrier needed for this private region)
#pragma unroll
    for (int v = 0; v < 8; v++) {
      pbuf[wave][rbase + v][n]      = (_Float16)p0[v];
      pbuf[wave][rbase + v][16 + n] = (_Float16)p1[v];
    }
    h16 pa;
    {
      int m  = lane & 15;
      int ko = (lane >= 16) ? 8 : 0;
      const _Float16* p = &pbuf[wave][m][ko];
      h8 c0 = *(const h8*)(p);
      h8 c1 = *(const h8*)(p + 16);
#pragma unroll
      for (int i = 0; i < 8; i++) { pa[i] = c0[i]; pa[8 + i] = c1[i]; }
    }

    // O += P @ V  (B operand from LDS vT tile: lane half h holds keys {8h..8h+7, 16+8h..16+8h+7})
#pragma unroll
    for (int nt = 0; nt < 4; nt++) {
      h16 bv = load_b_f16(&vlds[ibuf][nt * 16][0], 32, lane);
      acc[nt] = wmma32(pa, bv, acc[nt]);
    }
    __syncthreads();                     // done with LDS[ibuf] before reuse
  }

  // epilogue: out = acc / l
#pragma unroll
  for (int nt = 0; nt < 4; nt++) {
    int col = nt * 16 + n;
#pragma unroll
    for (int v = 0; v < 8; v++) {
      int row = rg0 + rbase + v;
      out[(size_t)row * DK + col] = acc[nt][v] / lrow[v];
    }
  }
}

// ---------------------------------------------------------------------------
// Host entry.  inputs: 0 Q, 1 K, 2 V, 3 Wq, 4 bq, 5 Wk, 6 bk, 7 Wv, 8 bv, 9 mask
// ---------------------------------------------------------------------------
extern "C" void kernel_launch(void* const* d_in, const int* in_sizes, int n_in,
                              void* d_out, int out_size, void* d_ws, size_t ws_size,
                              hipStream_t stream) {
  const float* Q  = (const float*)d_in[0];
  const float* K  = (const float*)d_in[1];
  const float* V  = (const float*)d_in[2];
  const float* Wq = (const float*)d_in[3];
  const float* bq = (const float*)d_in[4];
  const float* Wk = (const float*)d_in[5];
  const float* bk = (const float*)d_in[6];
  const float* Wv = (const float*)d_in[7];
  const float* bv = (const float*)d_in[8];
  float* out = (float*)d_out;

  const size_t WT_BYTES  = (size_t)3 * DK * DMODEL * 2;   // 393216
  const size_t ACT_BYTES = (size_t)ROWS * DK * 2;         // 2097152

  char* ws = (char*)d_ws;
  _Float16* WT   = (_Float16*)ws;
  _Float16* q16  = (_Float16*)(ws + WT_BYTES);
  _Float16* k16  = (_Float16*)(ws + WT_BYTES + ACT_BYTES);
  _Float16* vT16 = (_Float16*)(ws + WT_BYTES + 2 * ACT_BYTES);

  sha_wprep<<<dim3((DMODEL * DK + 255) / 256, 3), 256, 0, stream>>>(Wq, Wk, Wv, WT);

  const float inv_sqrt_dk = 0.125f;  // fold 1/sqrt(64) into q
  sha_proj<<<ROWS / 64, 128, 0, stream>>>(Q, WT + 0 * DK * DMODEL, bq,
                                          q16, (_Float16*)nullptr, 0, inv_sqrt_dk);
  sha_proj<<<ROWS / 64, 128, 0, stream>>>(K, WT + 1 * DK * DMODEL, bk,
                                          k16, (_Float16*)nullptr, 0, 1.0f);
  sha_proj<<<ROWS / 64, 128, 0, stream>>>(V, WT + 2 * DK * DMODEL, bv,
                                          (_Float16*)nullptr, vT16, 1, 1.0f);

  sha_attn<<<ROWS / 64, 128, 0, stream>>>(q16, k16, vT16, out);
}
